// MultiHeadQuasiLSTMCell_77730318123459
// MI455X (gfx1250) — matrix operand order repaired and checked by reading, not compiled, not run on hardware
//
#include <hip/hip_runtime.h>
#include <hip/hip_bf16.h>

// ---------------------------------------------------------------------------
// MultiHeadQuasiLSTM for MI455X (gfx1250, wave32, WMMA bf16 + async-to-LDS)
//   pack   : f32 -> bf16 (activations), f32 -> bf16 transposed (weights)
//   gates  = act(X @ W + b)   -> 4x WMMA GEMM (async LDS staging, dbl-buffer)
//   c,h    = sequential per-channel scan; h written as bf16
//   y      = h @ Wp + bp      -> 1x WMMA GEMM
// ---------------------------------------------------------------------------

typedef __attribute__((ext_vector_type(16))) __bf16 bf16x16;
typedef __attribute__((ext_vector_type(8)))  float  v8f;
typedef int vi4 __attribute__((vector_size(16)));     // 16B vector for async b128

#define B_   4
#define S_   2048
#define HD   1024              // H*D
#define M_   (B_ * S_)         // 8192
#define K_   1024              // HDIM (and inner dim of out-proj)
#define N_   HD                // 1024

#define BM 128
#define BN 128
#define BK 32
#define LSTR 40                // LDS row stride in ushorts (80B, 16B-aligned rows)

// ---------------- CDNA5 async-copy helpers (builtin if present, else asm) ----
#if defined(__has_builtin)
#  if __has_builtin(__builtin_amdgcn_global_load_async_to_lds_b128)
#    define ASYNC_HAS_BUILTIN 1
#  endif
#  if __has_builtin(__builtin_amdgcn_s_wait_asynccnt)
#    define WAITASYNC_HAS_BUILTIN 1
#  endif
#endif
#ifndef ASYNC_HAS_BUILTIN
#  define ASYNC_HAS_BUILTIN 0
#endif
#ifndef WAITASYNC_HAS_BUILTIN
#  define WAITASYNC_HAS_BUILTIN 0
#endif

__device__ __forceinline__ void async_copy_b128(const void* g, void* l) {
#if defined(__HIP_DEVICE_COMPILE__)
#if ASYNC_HAS_BUILTIN
    __builtin_amdgcn_global_load_async_to_lds_b128(
        (__attribute__((address_space(1))) vi4*)g,
        (__attribute__((address_space(3))) vi4*)l, 0, 0);
#else
    asm volatile("global_load_async_to_lds_b128 %0, %1, off"
                 :: "v"((unsigned)(unsigned long long)
                            (__attribute__((address_space(3))) char*)l),
                    "v"((unsigned long long)g)
                 : "memory");
#endif
#endif
}

template <int N>
__device__ __forceinline__ void wait_async() {
#if defined(__HIP_DEVICE_COMPILE__)
#if WAITASYNC_HAS_BUILTIN
    __builtin_amdgcn_s_wait_asynccnt(N);
#else
    asm volatile("s_wait_asynccnt %0" :: "i"(N) : "memory");
#endif
#endif
}

// ---------------------------------------------------------------------------
__device__ __forceinline__ unsigned short f2bf(float x) {
    unsigned u = __float_as_uint(x);
    unsigned r = (u + 0x7FFFu + ((u >> 16) & 1u)) >> 16;  // round-to-nearest-even
    return (unsigned short)r;
}

// Pack f32 -> bf16, 8 elements/thread, 16B stores.
__launch_bounds__(256)
__global__ void pack_bf16(const float* __restrict__ src,
                          unsigned short* __restrict__ dst, int n)
{
    int i = (blockIdx.x * 256 + threadIdx.x) * 8;
    if (i + 8 > n) return;
    float4 a = *(const float4*)(src + i);
    float4 b = *(const float4*)(src + i + 4);
    uint4 o;
    o.x = (unsigned)f2bf(a.x) | ((unsigned)f2bf(a.y) << 16);
    o.y = (unsigned)f2bf(a.z) | ((unsigned)f2bf(a.w) << 16);
    o.z = (unsigned)f2bf(b.x) | ((unsigned)f2bf(b.y) << 16);
    o.w = (unsigned)f2bf(b.z) | ((unsigned)f2bf(b.w) << 16);
    *(uint4*)(dst + i) = o;
}

// Pack-transpose weights: W (K x N f32, row-major) -> Wt (N x K bf16).
__launch_bounds__(256)
__global__ void packT_bf16(const float* __restrict__ W,
                           unsigned short* __restrict__ Wt)
{
    int idx = blockIdx.x * 256 + threadIdx.x;      // over K_*N_
    int k = idx / N_, n = idx % N_;
    Wt[(size_t)n * K_ + k] = f2bf(W[idx]);
}

// ---------------------------------------------------------------------------
enum { EPI_NONE = 0, EPI_SIGP1 = 1, EPI_SIG = 2, EPI_TANH = 3 };

// C = A (MxK bf16) x Bt^T (Bt is NxK bf16) + bias, epilogue-fused activation.
// 128x128 block tile, BK=32, 8 wave32s, double-buffered async LDS staging.
template <int EPI>
__launch_bounds__(256)
__global__ void gemm_bf16_async(const unsigned short* __restrict__ A,
                                const unsigned short* __restrict__ Bt,
                                const float* __restrict__ bias,
                                float* __restrict__ Out)
{
    __shared__ unsigned short lds[2 * 256 * LSTR];  // 2 bufs x (A 128 + B 128) rows

    const int tid  = threadIdx.x;
    const int lane = tid & 31;
    const int wave = tid >> 5;      // 0..7
    const int half = lane >> 4;     // 0/1
    const int l16  = lane & 15;

    const int n0 = blockIdx.x * BN;
    const int m0 = blockIdx.y * BM;
    const int wm = wave >> 1;       // 0..3 -> 32-row strip
    const int wn = wave & 1;        // 0..1 -> 64-col strip

    v8f acc[2][4];
#pragma unroll
    for (int i = 0; i < 2; ++i)
#pragma unroll
        for (int j = 0; j < 4; ++j) acc[i][j] = (v8f){0, 0, 0, 0, 0, 0, 0, 0};

    // issue one tile stage: 512 x 16B segments for A, 512 x 16B for B
    auto stage = [&](int b, int k0) {
        unsigned short* Al = lds + b * (256 * LSTR);
        unsigned short* Bl = Al + 128 * LSTR;
#pragma unroll
        for (int q = 0; q < 2; ++q) {
            int seg = tid + 256 * q;            // 0..511
            int row = seg >> 2, cs = seg & 3;   // tile row, cs = 8-elem chunk
            async_copy_b128(A  + (size_t)(m0 + row) * K_ + k0 + cs * 8,
                            Al + row * LSTR + cs * 8);
            async_copy_b128(Bt + (size_t)(n0 + row) * K_ + k0 + cs * 8,
                            Bl + row * LSTR + cs * 8);
        }
    };

    const int NIT = K_ / BK;   // 32
    stage(0, 0);
    for (int it = 0; it < NIT; ++it) {
        const int cur = it & 1;
        if (it + 1 < NIT) {
            stage(cur ^ 1, (it + 1) * BK);
            wait_async<4>();   // in-order: previous group of 4/wave complete
        } else {
            wait_async<0>();
        }
        __syncthreads();

        const unsigned short* Al = lds + cur * (256 * LSTR);
        const unsigned short* Bl = Al + 128 * LSTR;

        bf16x16 afrag[2], bfrag[4];
#pragma unroll
        for (int mi = 0; mi < 2; ++mi) {
            // A 16x32 frag: lane=M, VGPR0-3 = K 8*half..+7, VGPR4-7 = K 16+8*half..+7
            const int row = wm * 32 + mi * 16 + l16;
            union { uint4 q[2]; bf16x16 v; } u;
            u.q[0] = *(const uint4*)(Al + row * LSTR + 8 * half);
            u.q[1] = *(const uint4*)(Al + row * LSTR + 16 + 8 * half);
            afrag[mi] = u.v;
        }
#pragma unroll
        for (int ni = 0; ni < 4; ++ni) {
            // B 32x16 frag: lane=N, VGPR0-3 = K 8*half..+7, VGPR4-7 = K 16+8*half..+7
            const int n = wn * 64 + ni * 16 + l16;
            union { uint4 q[2]; bf16x16 v; } u;
            u.q[0] = *(const uint4*)(Bl + n * LSTR + 8 * half);
            u.q[1] = *(const uint4*)(Bl + n * LSTR + 16 + 8 * half);
            bfrag[ni] = u.v;
        }

#pragma unroll
        for (int mi = 0; mi < 2; ++mi)
#pragma unroll
            for (int ni = 0; ni < 4; ++ni)
                acc[mi][ni] = __builtin_amdgcn_wmma_f32_16x16x32_bf16(
                    false, afrag[mi], false, bfrag[ni],
                    (short)0, acc[mi][ni], false, false);

        __syncthreads();   // all waves done reading buf[cur] before next stage
    }

    // epilogue: bias + activation; C/D layout row = r + 8*half, col = l16
#pragma unroll
    for (int mi = 0; mi < 2; ++mi) {
#pragma unroll
        for (int ni = 0; ni < 4; ++ni) {
            const int colg = n0 + wn * 64 + ni * 16 + l16;
            const float bv = bias[colg];
#pragma unroll
            for (int r = 0; r < 8; ++r) {
                int rowg = m0 + wm * 32 + mi * 16 + r + half * 8;
                float x = acc[mi][ni][r] + bv;
                if (EPI == EPI_SIGP1)      x = 1.0f / (1.0f + __expf(-(x + 1.0f)));
                else if (EPI == EPI_SIG)   x = 1.0f / (1.0f + __expf(-x));
                else if (EPI == EPI_TANH)  x = tanhf(x);
                Out[(size_t)rowg * N_ + colg] = x;
            }
        }
    }
}

// ---------------------------------------------------------------------------
// Sequential per-channel scan over S; h emitted as bf16 for the output GEMM.
__launch_bounds__(256)
__global__ void qlstm_scan(const float* __restrict__ fa, const float* __restrict__ ia,
                           const float* __restrict__ za, const float* __restrict__ oa,
                           const float* __restrict__ c0,
                           unsigned short* __restrict__ hbuf,
                           float* __restrict__ lastc, float* __restrict__ lasth)
{
    int ch = blockIdx.x * blockDim.x + threadIdx.x;   // 0..4095
    if (ch >= B_ * HD) return;
    int b  = ch / HD;
    int hd = ch % HD;
    float c = c0[ch];
    float h = 0.0f;
    size_t base = (size_t)b * S_ * HD + hd;
    for (int t = 0; t < S_; ++t) {
        size_t idx = base + (size_t)t * HD;
        c = fa[idx] * c + ia[idx] * za[idx];
        h = oa[idx] * tanhf(c);
        hbuf[idx] = f2bf(h);
    }
    lastc[ch] = c;
    lasth[ch] = h;
}

// ---------------------------------------------------------------------------
extern "C" void kernel_launch(void* const* d_in, const int* in_sizes, int n_in,
                              void* d_out, int out_size, void* d_ws, size_t ws_size,
                              hipStream_t stream)
{
    const float* f_in = (const float*)d_in[0];
    const float* i_in = (const float*)d_in[1];
    const float* z_in = (const float*)d_in[2];
    const float* o_in = (const float*)d_in[3];
    const float* c0   = (const float*)d_in[4];
    // d_in[5] = h0 (unused by the reference computation)
    const float* Wf = (const float*)d_in[6];   const float* bfv = (const float*)d_in[7];
    const float* Wi = (const float*)d_in[8];   const float* biv = (const float*)d_in[9];
    const float* Wz = (const float*)d_in[10];  const float* bzv = (const float*)d_in[11];
    const float* Wo = (const float*)d_in[12];  const float* bov = (const float*)d_in[13];
    const float* Wp = (const float*)d_in[14];  const float* bpv = (const float*)d_in[15];

    float* y     = (float*)d_out;                    // (B,S,HDIM) = 8388608
    float* lastc = y + (size_t)M_ * N_;
    float* lasth = lastc + (size_t)B_ * HD;

    const size_t MK = (size_t)M_ * K_;               // 8388608
    const size_t NK = (size_t)N_ * K_;               // 1048576

    char* w = (char*)d_ws;
    unsigned short* aF = (unsigned short*)w;  w += MK * 2;
    unsigned short* aI = (unsigned short*)w;  w += MK * 2;
    unsigned short* aZ = (unsigned short*)w;  w += MK * 2;
    unsigned short* aO = (unsigned short*)w;  w += MK * 2;
    unsigned short* tF = (unsigned short*)w;  w += NK * 2;
    unsigned short* tI = (unsigned short*)w;  w += NK * 2;
    unsigned short* tZ = (unsigned short*)w;  w += NK * 2;
    unsigned short* tO = (unsigned short*)w;  w += NK * 2;
    unsigned short* tP = (unsigned short*)w;  w += NK * 2;
    float* gF = (float*)w;  w += MK * 4;
    float* gI = (float*)w;  w += MK * 4;
    float* gZ = (float*)w;  w += MK * 4;
    float* gO = (float*)w;  w += MK * 4;
    unsigned short* hb = (unsigned short*)w;  // MK * 2

    dim3 block(256);
    dim3 gpack((unsigned)(MK / (256 * 8)));          // 4096
    dim3 gpt((unsigned)(NK / 256));                  // 4096
    dim3 ggemm(N_ / BN, M_ / BM);                    // (8, 64)

    pack_bf16<<<gpack, block, 0, stream>>>(f_in, aF, (int)MK);
    pack_bf16<<<gpack, block, 0, stream>>>(i_in, aI, (int)MK);
    pack_bf16<<<gpack, block, 0, stream>>>(z_in, aZ, (int)MK);
    pack_bf16<<<gpack, block, 0, stream>>>(o_in, aO, (int)MK);
    packT_bf16<<<gpt, block, 0, stream>>>(Wf, tF);
    packT_bf16<<<gpt, block, 0, stream>>>(Wi, tI);
    packT_bf16<<<gpt, block, 0, stream>>>(Wz, tZ);
    packT_bf16<<<gpt, block, 0, stream>>>(Wo, tO);
    packT_bf16<<<gpt, block, 0, stream>>>(Wp, tP);

    gemm_bf16_async<EPI_SIGP1><<<ggemm, block, 0, stream>>>(aF, tF, bfv, gF);
    gemm_bf16_async<EPI_SIG  ><<<ggemm, block, 0, stream>>>(aI, tI, biv, gI);
    gemm_bf16_async<EPI_TANH ><<<ggemm, block, 0, stream>>>(aZ, tZ, bzv, gZ);
    gemm_bf16_async<EPI_SIG  ><<<ggemm, block, 0, stream>>>(aO, tO, bov, gO);

    qlstm_scan<<<dim3((B_ * HD + 255) / 256), block, 0, stream>>>(
        gF, gI, gZ, gO, c0, hb, lastc, lasth);

    gemm_bf16_async<EPI_NONE ><<<ggemm, block, 0, stream>>>(hb, tP, bpv, y);
}
